// DynamicFourierBlockSEKG_87162066305797
// MI455X (gfx1250) — matrix operand order repaired and checked by reading, not compiled, not run on hardware
//
#include <hip/hip_runtime.h>
#include <math.h>

// ---------------- problem constants ----------------
#define BB    2
#define CCH   64        // DIM
#define HH    256
#define WWp   256
#define WF    129       // W/2+1
#define PF    (HH*WF)   // 33024  freq-domain pixels
#define PP    (HH*WWp)  // 65536  spatial pixels
#define ROWS  (BB*CCH*HH) // 32768
#define NP1   272       // 258 (129 re + 129 im) padded to 17 wmma tiles
#define KW2   288       // 258 padded to 9 k-chunks of 32
#define FGNH  32
#define FFNH  128
#define HEADM 1152      // 2*64*9

typedef __attribute__((ext_vector_type(16))) _Float16 v16h;
typedef __attribute__((ext_vector_type(8)))  float    v8f;

// ---------------- WMMA fragment loaders (contiguous k order for A and B, wave32) ----------
// A 16x32 f16: lane L: row = L&15; khalf = L>>4;
//   element e -> K = 16*khalf + e
__device__ __forceinline__ v16h load_a16(const _Float16* __restrict__ Ar, int hk) {
  v16h a;
#pragma unroll
  for (int e = 0; e < 16; ++e) {
    int kk = hk * 16 + e;
    a[e] = Ar[kk];
  }
  return a;
}
// B 32x16 f16: lane L: col = L&15; element e -> K = 16*(L>>4) + e (stride ldb rows)
__device__ __forceinline__ v16h load_b16(const _Float16* __restrict__ Bp, size_t ldb) {
  v16h b;
#pragma unroll
  for (int e = 0; e < 16; ++e) b[e] = Bp[(size_t)e * ldb];
  return b;
}

// ---------------- generic batched WMMA GEMM (16M x 64N per wave) ----------------
// C[z][m][n] = sum_k A[m][k] * B[z][k][n]  (+bias[m]) (+res[z][m][n])
// M,N multiples of 16; K multiple of 32.  8 waves/block; each wave owns 4 n-tiles
// and reuses one A fragment per K-chunk across 4 back-to-back WMMAs.
template <bool BIAS, bool RES, bool STOREH>
__global__ __launch_bounds__(256) void gemm_wmma_f16(
    const _Float16* __restrict__ A, int lda,
    const _Float16* __restrict__ B, int ldb, long long sB,
    void* __restrict__ C, int ldc, long long sC,
    const float* __restrict__ bias,
    const float* __restrict__ res, long long sR,
    int N, int K) {
  int wave = threadIdx.x >> 5;
  int lane = threadIdx.x & 31;
  int col  = lane & 15;
  int hk   = lane >> 4;
  int ntiles = (N + 15) >> 4;
  int nt0 = (blockIdx.x * 8 + wave) * 4;     // first of this wave's 4 n-tiles
  if (nt0 >= ntiles) return;                 // wave-uniform: EXEC stays all-ones
  int nvalid = ntiles - nt0; if (nvalid > 4) nvalid = 4;
  int m0 = blockIdx.y * 16;
  int z  = blockIdx.z;
  const _Float16* Bz = B + (size_t)z * sB;
  const _Float16* Ar = A + (size_t)(m0 + col) * lda;   // row = lane&15 for A too
  v8f acc0 = {}, acc1 = {}, acc2 = {}, acc3 = {};
  for (int k0 = 0; k0 < K; k0 += 32) {
    v16h av = load_a16(Ar + k0, hk);
    const _Float16* Bp = Bz + (size_t)(k0 + hk * 16) * ldb + nt0 * 16 + col;
    if (k0 + 32 < K) __builtin_prefetch(Bp + (size_t)32 * ldb, 0, 1);
    {
      v16h bv = load_b16(Bp, (size_t)ldb);
      acc0 = __builtin_amdgcn_wmma_f32_16x16x32_f16(false, av, false, bv, (short)0, acc0, false, false);
    }
    if (nvalid > 1) {
      v16h bv = load_b16(Bp + 16, (size_t)ldb);
      acc1 = __builtin_amdgcn_wmma_f32_16x16x32_f16(false, av, false, bv, (short)0, acc1, false, false);
    }
    if (nvalid > 2) {
      v16h bv = load_b16(Bp + 32, (size_t)ldb);
      acc2 = __builtin_amdgcn_wmma_f32_16x16x32_f16(false, av, false, bv, (short)0, acc2, false, false);
    }
    if (nvalid > 3) {
      v16h bv = load_b16(Bp + 48, (size_t)ldb);
      acc3 = __builtin_amdgcn_wmma_f32_16x16x32_f16(false, av, false, bv, (short)0, acc3, false, false);
    }
  }
  // D layout: m = 8*(lane>>4)+v, n = lane&15
#pragma unroll
  for (int t = 0; t < 4; ++t) {
    if (t >= nvalid) break;
    v8f acc = (t == 0) ? acc0 : (t == 1) ? acc1 : (t == 2) ? acc2 : acc3;
#pragma unroll
    for (int v = 0; v < 8; ++v) {
      int m = m0 + hk * 8 + v;
      int n = (nt0 + t) * 16 + col;
      float val = acc[v];
      if (BIAS) val += bias[m];
      if (RES)  val += res[(size_t)z * sR + (size_t)m * ldc + n];
      if (STOREH) ((_Float16*)C)[(size_t)z * sC + (size_t)m * ldc + n] = (_Float16)val;
      else        ((float*)C)[(size_t)z * sC + (size_t)m * ldc + n] = val;
    }
  }
}

// ---------------- small elementwise / setup kernels ----------------
__global__ void f2h_k(const float* __restrict__ in, _Float16* __restrict__ o, int n) {
  int i = blockIdx.x * 256 + threadIdx.x;
  if (i < n) o[i] = (_Float16)in[i];
}

// Row-DFT matrix Bw1[w][n] (256 x 272): n<129 -> cos, 129..257 -> -sin, else 0 (ortho 1/16)
__global__ void gen_bw1(_Float16* __restrict__ o) {
  int i = blockIdx.x * 256 + threadIdx.x;
  if (i >= 256 * NP1) return;
  int w = i / NP1, n = i % NP1;
  float v = 0.f;
  if (n < 129)       v =  cosf(6.2831853071795864f * (float)((w * n) & 255) * (1.f / 256.f));
  else if (n < 258)  v = -sinf(6.2831853071795864f * (float)((w * (n - 129)) & 255) * (1.f / 256.f));
  o[i] = (_Float16)(v * 0.0625f);
}
// Column DFT matrices T[h'][h], sign=-1 fwd, +1 inv (ortho 1/16)
__global__ void gen_t(_Float16* __restrict__ tr, _Float16* __restrict__ ti, float sgn) {
  int i = blockIdx.x * 256 + threadIdx.x;
  if (i >= 256 * 256) return;
  int a = i >> 8, b = i & 255;
  float ang = 6.2831853071795864f * (float)((a * b) & 255) * (1.f / 256.f);
  tr[i] = (_Float16)(cosf(ang) * 0.0625f);
  ti[i] = (_Float16)(sgn * sinf(ang) * 0.0625f);
}
// irfft matrix Bw2[k2][w] (288 x 256): k2<129: a_k cos; 129..257: -a_k sin; pad rows 0
__global__ void gen_bw2(_Float16* __restrict__ o) {
  int i = blockIdx.x * 256 + threadIdx.x;
  if (i >= KW2 * 256) return;
  int k2 = i / 256, w = i % 256;
  float v = 0.f;
  if (k2 < 258) {
    int k = (k2 < 129) ? k2 : k2 - 129;
    float a = (k == 0 || k == 128) ? 1.f : 2.f;
    float ang = 6.2831853071795864f * (float)((k * w) & 255) * (1.f / 256.f);
    v = (k2 < 129) ? a * cosf(ang) : -a * sinf(ang);
  }
  o[i] = (_Float16)(v * 0.0625f);
}

// LayerNorm over C=64 at each (b,pixel); writes f16 (B,64,P)
__global__ void layernorm64(const float* __restrict__ x, const float* __restrict__ g,
                            const float* __restrict__ bt, _Float16* __restrict__ oh, int P) {
  int idx = blockIdx.x * 256 + threadIdx.x;
  if (idx >= BB * P) return;
  int b = idx / P, pix = idx % P;
  const float* xb = x + (size_t)b * CCH * P + pix;
  float s = 0.f, s2 = 0.f;
  for (int c = 0; c < CCH; ++c) { float v = xb[(size_t)c * P]; s += v; s2 += v * v; }
  float m = s * (1.f / CCH);
  float inv = rsqrtf(s2 * (1.f / CCH) - m * m + 1e-5f);
  _Float16* ob = oh + (size_t)b * CCH * P + pix;
  for (int c = 0; c < CCH; ++c) {
    float v = xb[(size_t)c * P];
    ob[(size_t)c * P] = (_Float16)(((v - m) * inv) * g[c] + bt[c]);
  }
}

// Zr=P.r-Q.i, Zi=P.i+Q.r -> mag/phase (f32) and fgn_in (f16, mag ch 0..63, phase 64..127)
__global__ void combine_fwd(const float* __restrict__ P, const float* __restrict__ Q,
                            float* __restrict__ mag, float* __restrict__ ph,
                            _Float16* __restrict__ fgn) {
  int idx = blockIdx.x * 256 + threadIdx.x;
  if (idx >= BB * CCH * HH * WF) return;
  int k = idx % WF; int t = idx / WF; int h = t % HH; int bc = t / HH;
  size_t base = (size_t)h * (128 * NP1) + (size_t)bc * NP1;
  float pr = P[base + k], pi = P[base + 129 + k];
  float qr = Q[base + k], qi = Q[base + 129 + k];
  float zr = pr - qi, zi = pi + qr;
  float mv = sqrtf(zr * zr + zi * zi) + 1e-8f;
  float pv = atan2f(zi, zr);
  size_t mi = ((size_t)bc * HH + h) * WF + k;
  mag[mi] = mv; ph[mi] = pv;
  int b = bc >> 6, c = bc & 63;
  size_t p = (size_t)h * WF + k;
  fgn[((size_t)b * 128 + c) * PF + p]      = (_Float16)mv;
  fgn[((size_t)b * 128 + 64 + c) * PF + p] = (_Float16)pv;
}

// depthwise 3x3 SAME + hardswish
__global__ void dwconv3_k(const float* __restrict__ in, const float* __restrict__ w,
                          float* __restrict__ out, int CH, int H, int W, int total) {
  int idx = blockIdx.x * 256 + threadIdx.x;
  if (idx >= total) return;
  int ww = idx % W; int t = idx / W; int h = t % H; int bc = t / H; int ch = bc % CH;
  const float* wp = w + ch * 9;
  const float* ib = in + (size_t)bc * H * W;
  float acc = 0.f;
#pragma unroll
  for (int di = 0; di < 3; ++di)
#pragma unroll
    for (int dj = 0; dj < 3; ++dj) {
      int hh = h + di - 1, wk = ww + dj - 1;
      if (hh >= 0 && hh < H && wk >= 0 && wk < W) acc += ib[hh * W + wk] * wp[di * 3 + dj];
    }
  out[idx] = acc * fminf(fmaxf(acc + 3.f, 0.f), 6.f) * (1.f / 6.f);  // hardswish
}

__global__ void meanpool_k(const float* __restrict__ in, float* __restrict__ out, int P) {
  __shared__ float red[256];
  int bc = blockIdx.x;
  const float* p = in + (size_t)bc * P;
  float s = 0.f;
  for (int i = threadIdx.x; i < P; i += 256) s += p[i];
  red[threadIdx.x] = s; __syncthreads();
  for (int o = 128; o > 0; o >>= 1) { if (threadIdx.x < o) red[threadIdx.x] += red[threadIdx.x + o]; __syncthreads(); }
  if (threadIdx.x == 0) out[bc] = red[0] / (float)P;
}

__global__ void gate_k(const float* __restrict__ pooled, const float* __restrict__ wgc,
                       const float* __restrict__ bgc, float* __restrict__ gate) {
  int i = threadIdx.x;
  if (i < 64) {
    int b = i >> 5, o = i & 31;
    float s = bgc[o];
    for (int c = 0; c < 32; ++c) s += wgc[o * 32 + c] * pooled[b * 32 + c];
    gate[i] = 1.f / (1.f + expf(-s));
  }
}

__global__ void se_k(const float* __restrict__ pooled2, const float* __restrict__ w1,
                     const float* __restrict__ b1, const float* __restrict__ w2,
                     const float* __restrict__ b2, float* __restrict__ s2) {
  __shared__ float s1[64];
  int i = threadIdx.x;
  if (i < 64) {
    int b = i >> 5, r = i & 31;
    float s = b1[r];
    for (int c = 0; c < 128; ++c) s += w1[r * 128 + c] * pooled2[b * 128 + c];
    s1[i] = fmaxf(s, 0.f);
  }
  __syncthreads();
  int b = i >> 7, o = i & 127;
  float s = b2[o];
  for (int r = 0; r < 32; ++r) s += w2[o * 32 + r] * s1[b * 32 + r];
  s2[i] = fminf(fmaxf(s + 3.f, 0.f), 6.f) * (1.f / 6.f);  // hardsigmoid
}

__global__ void scale_half_k(const float* __restrict__ in, const float* __restrict__ sc,
                             _Float16* __restrict__ o, int P, int n) {
  int i = blockIdx.x * 256 + threadIdx.x;
  if (i < n) o[i] = (_Float16)(in[i] * sc[i / P]);
}

// ---------------- fused head-GEMM + softmax(9) + dynamic filter + polar->rect -------
// grid (PF/16, B), 256 thr. LDS holds 576x16 logits (36KB) + 64x16 fm (4KB).
__global__ __launch_bounds__(256) void fgn_head_fused(
    const _Float16* __restrict__ Wh, const float* __restrict__ bias,
    const _Float16* __restrict__ lg, const float* __restrict__ mag,
    const float* __restrict__ phase, _Float16* __restrict__ Fbuf) {
  __shared__ float fil[576 * 16];
  __shared__ float fmb[64 * 16];
  int b = blockIdx.y;
  int p0 = blockIdx.x * 16;
  int wave = threadIdx.x >> 5, lane = threadIdx.x & 31;
  int col = lane & 15, hk = lane >> 4;
  // B fragment (32 x 16 of gated activations) loaded once, reused by 72 WMMAs
  v16h bt;
  const _Float16* lgp = lg + (size_t)b * 32 * PF + p0 + col;
#pragma unroll
  for (int e = 0; e < 16; ++e) bt[e] = lgp[(size_t)(hk * 16 + e) * PF];

  for (int pass = 0; pass < 2; ++pass) {   // 0: mag filters, 1: phase filters
    int base = pass * 576;
    for (int t = wave; t < 36; t += 8) {   // 36 M-tiles of 16 output channels
      v16h av = load_a16(Wh + (size_t)(base + t * 16 + col) * 32, hk);
      v8f acc = {};
      acc = __builtin_amdgcn_wmma_f32_16x16x32_f16(false, av, false, bt,
                                                   (short)0, acc, false, false);
#pragma unroll
      for (int v = 0; v < 8; ++v) {
        int m = t * 16 + hk * 8 + v;
        fil[m * 16 + col] = acc[v] + bias[base + m];
      }
    }
    __syncthreads();
    const float* src = pass ? phase : mag;
    for (int it = 0; it < 4; ++it) {
      int idx = threadIdx.x + it * 256;     // (c in [0,64)) x (pc in [0,16))
      int c = idx >> 4, pc = idx & 15;
      int p = p0 + pc;
      int h = p / WF, wk = p % WF;
      float w9[9]; float mx = -1e30f;
#pragma unroll
      for (int k2 = 0; k2 < 9; ++k2) { w9[k2] = fil[(c * 9 + k2) * 16 + pc]; mx = fmaxf(mx, w9[k2]); }
      float ssum = 0.f;
#pragma unroll
      for (int k2 = 0; k2 < 9; ++k2) { w9[k2] = expf(w9[k2] - mx); ssum += w9[k2]; }
      float inv = 1.f / ssum;
      const float* plane = src + (size_t)(b * 64 + c) * HH * WF;
      float accv = 0.f;
#pragma unroll
      for (int k2 = 0; k2 < 9; ++k2) {
        int hh2 = h + k2 / 3 - 1, wk2 = wk + k2 % 3 - 1;
        float v = 0.f;
        if (hh2 >= 0 && hh2 < HH && wk2 >= 0 && wk2 < WF) v = plane[hh2 * WF + wk2];
        accv += w9[k2] * inv * v;
      }
      if (pass == 0) fmb[c * 16 + pc] = accv;
      else {
        float fm = fmb[c * 16 + pc];
        size_t rowb = ((size_t)(b * 64 + c) * HH + h) * NP1;
        Fbuf[rowb + wk]       = (_Float16)(fm * cosf(accv));
        Fbuf[rowb + 129 + wk] = (_Float16)(fm * sinf(accv));
      }
    }
    __syncthreads();
  }
}

// Gr=P.r-Q.i, Gi=P.i+Q.r packed [row][288] f16 for irfft-W GEMM
__global__ void combine_inv(const float* __restrict__ P, const float* __restrict__ Q,
                            _Float16* __restrict__ G) {
  int idx = blockIdx.x * 256 + threadIdx.x;
  if (idx >= BB * CCH * HH * WF) return;
  int k = idx % WF; int t = idx / WF; int h = t % HH; int bc = t / HH;
  size_t base = (size_t)h * (128 * NP1) + (size_t)bc * NP1;
  float gr = P[base + k] - Q[base + 129 + k];
  float gi = P[base + 129 + k] + Q[base + k];
  size_t rb = ((size_t)bc * HH + h) * KW2;
  G[rb + k] = (_Float16)gr;
  G[rb + 129 + k] = (_Float16)gi;
}

// ---------------- host-side orchestration ----------------
extern "C" void kernel_launch(void* const* d_in, const int* in_sizes, int n_in,
                              void* d_out, int out_size, void* d_ws, size_t ws_size,
                              hipStream_t stream) {
  (void)in_sizes; (void)n_in; (void)out_size; (void)ws_size;
  const float* x     = (const float*)d_in[0];
  const float* n1g   = (const float*)d_in[1];
  const float* n1b   = (const float*)d_in[2];
  const float* w_l1  = (const float*)d_in[3];
  const float* w_fdw = (const float*)d_in[4];
  const float* w_gc  = (const float*)d_in[5];
  const float* b_gc  = (const float*)d_in[6];
  const float* w_hd  = (const float*)d_in[7];
  const float* b_hd  = (const float*)d_in[8];
  const float* n2g   = (const float*)d_in[9];
  const float* n2b   = (const float*)d_in[10];
  const float* w_f1  = (const float*)d_in[11];
  const float* w_fdw2= (const float*)d_in[12];
  const float* w_se1 = (const float*)d_in[13];
  const float* b_se1 = (const float*)d_in[14];
  const float* w_se2 = (const float*)d_in[15];
  const float* b_se2 = (const float*)d_in[16];
  const float* w_f2  = (const float*)d_in[17];
  const float* b_f2  = (const float*)d_in[18];
  float* out = (float*)d_out;

  // workspace carve (deterministic; ~433 MB)
  char* wp = (char*)d_ws;
  auto alloc = [&](size_t bytes) { void* p = wp; wp += (bytes + 255) & ~(size_t)255; return p; };
  _Float16* xn_h   = (_Float16*)alloc((size_t)ROWS * 256 * 2);
  _Float16* Y      = (_Float16*)alloc((size_t)ROWS * NP1 * 2);
  float*    Pbuf   = (float*)alloc((size_t)256 * 128 * NP1 * 4);
  float*    Qbuf   = (float*)alloc((size_t)256 * 128 * NP1 * 4);
  float*    mag    = (float*)alloc((size_t)BB * CCH * HH * WF * 4);
  float*    phase  = (float*)alloc((size_t)BB * CCH * HH * WF * 4);
  _Float16* fgn_h  = (_Float16*)alloc((size_t)BB * 128 * PF * 2);
  float*    local1 = (float*)alloc((size_t)BB * FGNH * PF * 4);
  float*    local2 = (float*)alloc((size_t)BB * FGNH * PF * 4);
  float*    pooled = (float*)alloc(BB * FGNH * 4);
  float*    gate   = (float*)alloc(BB * FGNH * 4);
  _Float16* lg_h   = (_Float16*)alloc((size_t)BB * FGNH * PF * 2);
  _Float16* Fbuf   = (_Float16*)alloc((size_t)ROWS * NP1 * 2);
  _Float16* Ghalf  = (_Float16*)alloc((size_t)ROWS * KW2 * 2);
  float*    X1     = (float*)alloc((size_t)BB * CCH * PP * 4);
  _Float16* xn2_h  = (_Float16*)alloc((size_t)BB * CCH * PP * 2);
  float*    yffn   = (float*)alloc((size_t)BB * FFNH * PP * 4);
  float*    y2     = (float*)alloc((size_t)BB * FFNH * PP * 4);
  _Float16* y2g_h  = (_Float16*)alloc((size_t)BB * FFNH * PP * 2);
  float*    pooled2= (float*)alloc(BB * FFNH * 4);
  float*    s2     = (float*)alloc(BB * FFNH * 4);
  _Float16* wl1_h  = (_Float16*)alloc(FGNH * 128 * 2);
  _Float16* whd_h  = (_Float16*)alloc(HEADM * 32 * 2);
  _Float16* wf1_h  = (_Float16*)alloc(FFNH * CCH * 2);
  _Float16* wf2_h  = (_Float16*)alloc(CCH * FFNH * 2);
  _Float16* Bw1    = (_Float16*)alloc(256 * NP1 * 2);
  _Float16* Tfr    = (_Float16*)alloc(256 * 256 * 2);
  _Float16* Tfi    = (_Float16*)alloc(256 * 256 * 2);
  _Float16* Tir    = (_Float16*)alloc(256 * 256 * 2);
  _Float16* Tii    = (_Float16*)alloc(256 * 256 * 2);
  _Float16* Bw2    = (_Float16*)alloc((size_t)KW2 * 256 * 2);

  auto blks = [](int n) { return (n + 255) / 256; };
  auto gxw  = [](int N) { return (((N + 15) / 16) + 31) / 32; };  // 8 waves * 4 tiles

  // weight f16 staging + DFT matrices (regenerated every call: deterministic)
  f2h_k<<<blks(FGNH * 128), 256, 0, stream>>>(w_l1, wl1_h, FGNH * 128);
  f2h_k<<<blks(HEADM * 32), 256, 0, stream>>>(w_hd, whd_h, HEADM * 32);
  f2h_k<<<blks(FFNH * CCH), 256, 0, stream>>>(w_f1, wf1_h, FFNH * CCH);
  f2h_k<<<blks(CCH * FFNH), 256, 0, stream>>>(w_f2, wf2_h, CCH * FFNH);
  gen_bw1<<<blks(256 * NP1), 256, 0, stream>>>(Bw1);
  gen_t<<<blks(256 * 256), 256, 0, stream>>>(Tfr, Tfi, -1.f);
  gen_t<<<blks(256 * 256), 256, 0, stream>>>(Tir, Tii, 1.f);
  gen_bw2<<<blks(KW2 * 256), 256, 0, stream>>>(Bw2);

  // ---- Fourier branch ----
  layernorm64<<<blks(BB * PP), 256, 0, stream>>>(x, n1g, n1b, xn_h, PP);

  // row rfft as GEMM: [32768 x 256] x [256 x 272] -> Y (f16)
  { dim3 g(gxw(NP1), ROWS / 16, 1);
    gemm_wmma_f16<false, false, true><<<g, 256, 0, stream>>>(
        xn_h, 256, Bw1, NP1, 0, Y, NP1, 0, nullptr, nullptr, 0, NP1, 256); }

  // column FFT: P=Tfr*Y, Q=Tfi*Y (batched over bc=128)
  { dim3 g(gxw(NP1), 16, 128);
    gemm_wmma_f16<false, false, false><<<g, 256, 0, stream>>>(
        Tfr, 256, Y, NP1, (long long)256 * NP1, Pbuf, 128 * NP1, NP1, nullptr, nullptr, 0, NP1, 256);
    gemm_wmma_f16<false, false, false><<<g, 256, 0, stream>>>(
        Tfi, 256, Y, NP1, (long long)256 * NP1, Qbuf, 128 * NP1, NP1, nullptr, nullptr, 0, NP1, 256); }

  combine_fwd<<<blks(BB * CCH * HH * WF), 256, 0, stream>>>(Pbuf, Qbuf, mag, phase, fgn_h);

  // FGN: conv1x1 128->32 (per batch), dw3x3+hardswish, gate, scale
  { dim3 g(gxw(PF), FGNH / 16, BB);
    gemm_wmma_f16<false, false, false><<<g, 256, 0, stream>>>(
        wl1_h, 128, fgn_h, PF, (long long)128 * PF, local1, PF, (long long)FGNH * PF,
        nullptr, nullptr, 0, PF, 128); }
  dwconv3_k<<<blks(BB * FGNH * PF), 256, 0, stream>>>(local1, w_fdw, local2, FGNH, HH, WF, BB * FGNH * PF);
  meanpool_k<<<BB * FGNH, 256, 0, stream>>>(local2, pooled, PF);
  gate_k<<<1, 64, 0, stream>>>(pooled, w_gc, b_gc, gate);
  scale_half_k<<<blks(BB * FGNH * PF), 256, 0, stream>>>(local2, gate, lg_h, PF, BB * FGNH * PF);

  // fused: head GEMM (K=32) + softmax(9) + dynamic filter + fm*(cos,sin) -> Fbuf
  { dim3 g(PF / 16, BB);
    fgn_head_fused<<<g, 256, 0, stream>>>(whd_h, b_hd, lg_h, mag, phase, Fbuf); }

  // inverse column FFT: P2=Tir*F, Q2=Tii*F (reuse P/Q)
  { dim3 g(gxw(NP1), 16, 128);
    gemm_wmma_f16<false, false, false><<<g, 256, 0, stream>>>(
        Tir, 256, Fbuf, NP1, (long long)256 * NP1, Pbuf, 128 * NP1, NP1, nullptr, nullptr, 0, NP1, 256);
    gemm_wmma_f16<false, false, false><<<g, 256, 0, stream>>>(
        Tii, 256, Fbuf, NP1, (long long)256 * NP1, Qbuf, 128 * NP1, NP1, nullptr, nullptr, 0, NP1, 256); }
  combine_inv<<<blks(BB * CCH * HH * WF), 256, 0, stream>>>(Pbuf, Qbuf, Ghalf);

  // irfft along W as GEMM + residual x  -> X1 = x + fft_out
  { dim3 g(gxw(256), ROWS / 16, 1);
    gemm_wmma_f16<false, true, false><<<g, 256, 0, stream>>>(
        Ghalf, KW2, Bw2, 256, 0, X1, 256, 0, nullptr, x, 0, 256, KW2); }

  // ---- FFN branch ----
  layernorm64<<<blks(BB * PP), 256, 0, stream>>>(X1, n2g, n2b, xn2_h, PP);
  { dim3 g(gxw(PP), FFNH / 16, BB);
    gemm_wmma_f16<false, false, false><<<g, 256, 0, stream>>>(
        wf1_h, CCH, xn2_h, PP, (long long)CCH * PP, yffn, PP, (long long)FFNH * PP,
        nullptr, nullptr, 0, PP, CCH); }
  dwconv3_k<<<blks(BB * FFNH * PP), 256, 0, stream>>>(yffn, w_fdw2, y2, FFNH, HH, WWp, BB * FFNH * PP);
  meanpool_k<<<BB * FFNH, 256, 0, stream>>>(y2, pooled2, PP);
  se_k<<<1, 256, 0, stream>>>(pooled2, w_se1, b_se1, w_se2, b_se2, s2);
  scale_half_k<<<blks(BB * FFNH * PP), 256, 0, stream>>>(y2, s2, y2g_h, PP, BB * FFNH * PP);
  { dim3 g(gxw(PP), CCH / 16, BB);
    gemm_wmma_f16<true, true, false><<<g, 256, 0, stream>>>(
        wf2_h, FFNH, y2g_h, PP, (long long)FFNH * PP, out, PP, (long long)CCH * PP,
        b_f2, X1, (long long)CCH * PP, PP, FFNH); }
}
